// AttentionModel_80633716015663
// MI455X (gfx1250) — matrix operand order repaired and checked by reading, not compiled, not run on hardware
//
#include <hip/hip_runtime.h>

typedef __bf16 bf16;
typedef bf16  bf16x16 __attribute__((ext_vector_type(16)));
typedef bf16  bf16x8  __attribute__((ext_vector_type(8)));
typedef float floatx8 __attribute__((ext_vector_type(8)));

#define NB   1024
#define NTOK 131
#define MR   (NB * NTOK)     // 134144 rows, multiple of 16 and 128
#define MPAD (MR + 32)       // pad rows so attention tile overreads stay in-bounds
#define DD   128
#define FFH  512
static constexpr float INV_SQRT_D = 0.08838834764831845f;

// ---------------------------------------------------------------- WMMA helpers

__device__ __forceinline__ floatx8 wmma_acc(bf16x16 a, bf16x16 b, floatx8 c) {
  return __builtin_amdgcn_wmma_f32_16x16x32_bf16(false, a, false, b, (short)0, c,
                                                 false, false);
}

// A fragment: 16x32 bf16 tile (row-major, leading dim ld). contiguous K order used by this file:
// lane<16 -> row=lane, K = 0..15; lane>=16 -> row=lane-16, K = 16..31
__device__ __forceinline__ bf16x16 load_a(const bf16* tile, int ld, int lane) {
  int row  = lane & 15;
  int koff = (lane & 16) ? 16 : 0;
  const bf16* p = tile + row * ld + koff;
  bf16x8 lo = *(const bf16x8*)p;
  bf16x8 hi = *(const bf16x8*)(p + 8);
  bf16x16 r;
#pragma unroll
  for (int i = 0; i < 8; ++i) { r[i] = lo[i]; r[i + 8] = hi[i]; }
  return r;
}

// Same, reading f32 source (LDS attention matrix), converting to bf16
__device__ __forceinline__ bf16x16 load_a_f32(const float* tile, int ld, int lane) {
  int row  = lane & 15;
  int koff = (lane & 16) ? 16 : 0;
  const float* p = tile + row * ld + koff;
  bf16x16 r;
#pragma unroll
  for (int i = 0; i < 8; ++i) { r[i] = (bf16)p[i]; r[i + 8] = (bf16)p[i + 8]; }
  return r;
}

// B fragment: 32x16 tile of row-major W (K x N), ld = N. contiguous K order (same as load_a):
// lane<16 -> col=lane, K=0..15; lane>=16 -> col=lane-16, K=16..31 (consecutive K per lane)
__device__ __forceinline__ bf16x16 load_b(const bf16* tile, int ld, int lane) {
  int col = lane & 15;
  int k0  = (lane & 16) ? 16 : 0;
  const bf16* p = tile + k0 * ld + col;
  bf16x16 r;
#pragma unroll
  for (int i = 0; i < 16; ++i) r[i] = p[i * ld];
  return r;
}

// B fragment where logical B = X^T: B[k][n] = X[n][k]; X row-major, leading dim ld.
// Per-lane data is 16 *contiguous* bf16 -> two 16B loads.
__device__ __forceinline__ bf16x16 load_bt(const bf16* tile, int ld, int lane) {
  int col = lane & 15;
  int k0  = (lane & 16) ? 16 : 0;
  const bf16* p = tile + col * ld + k0;
  bf16x8 lo = *(const bf16x8*)p;
  bf16x8 hi = *(const bf16x8*)(p + 8);
  bf16x16 r;
#pragma unroll
  for (int i = 0; i < 8; ++i) { r[i] = lo[i]; r[i + 8] = hi[i]; }
  return r;
}

// ---------------------------------------------------------------- weight convert
// Generalized: supports re-laying out W into column-concatenated matrices
// (dst[r*dstLd + colOff + c] = src[r*cols + c]).

struct CvtJob  { const float* src; bf16* dst; int rows, cols, dstLd, colOff; };
struct CvtJobs { CvtJob j[13]; };

__global__ void cvt_kernel(CvtJobs jobs) {
  int stride = gridDim.x * blockDim.x;
  int tid = blockIdx.x * blockDim.x + threadIdx.x;
  for (int k = 0; k < 13; ++k) {
    const float* s = jobs.j[k].src;
    bf16* d = jobs.j[k].dst;
    int cols = jobs.j[k].cols, dstLd = jobs.j[k].dstLd, colOff = jobs.j[k].colOff;
    int n = jobs.j[k].rows * cols;
    for (int i = tid; i < n; i += stride) {
      int r = i / cols, c = i - r * cols;
      d[r * dstLd + colOff + c] = (bf16)s[i];
    }
  }
}

// ---------------------------------------------------------------- embeddings

struct EmbedW {
  const float *iw1, *ib1, *iw2, *ib2;
  const float *lw1, *lb1, *lw2, *lb2;
  const float *nw1, *nb1, *nw2, *nb2;
};

__global__ void embed_kernel(const float* __restrict__ inp, EmbedW w,
                             float* __restrict__ h, bf16* __restrict__ hb) {
  long long tok = blockIdx.x;
  int n = (int)(tok % NTOK);
  const float *w1, *b1, *w2, *b2;
  int fi;
  if (n < 80)       { w1 = w.iw1; b1 = w.ib1; w2 = w.iw2; b2 = w.ib2; fi = 6; }
  else if (n < 130) { w1 = w.lw1; b1 = w.lb1; w2 = w.lw2; b2 = w.lb2; fi = 8; }
  else              { w1 = w.nw1; b1 = w.nb1; w2 = w.nw2; b2 = w.nb2; fi = 6; }
  __shared__ float x[8];
  __shared__ float hid[32];
  if (threadIdx.x < 8) x[threadIdx.x] = inp[tok * 9 + threadIdx.x];
  __syncthreads();
  if (threadIdx.x < 32) {
    float s = b1[threadIdx.x];
    for (int i = 0; i < fi; ++i) s += x[i] * w1[i * 32 + threadIdx.x];
    hid[threadIdx.x] = (s > 0.f) ? s : 0.01f * s;   // leaky relu
  }
  __syncthreads();
  int d = threadIdx.x;   // blockDim = 128
  float s = b2[d];
#pragma unroll 8
  for (int j = 0; j < 32; ++j) s += hid[j] * w2[j * DD + d];
  h[tok * DD + d]  = s;
  hb[tok * DD + d] = (bf16)s;
}

// ---------------------------------------------------------------- fused QKV GEMM
// Wqkv is column-concatenated [128][384] = [wq | wk | wv]. Each wave owns one
// 16-col tile of each output; A fragments are loaded once per M-tile and feed
// 12 WMMAs (3 accumulators). launch_bounds(256,1): keep all 12 B fragments
// register-resident (no scratch spills in the hot loop).

__global__ void __launch_bounds__(256, 1)
gemm_qkv(const bf16* __restrict__ A, const bf16* __restrict__ Wqkv,
         bf16* __restrict__ q, bf16* __restrict__ kk, bf16* __restrict__ v) {
  int lane = threadIdx.x & 31;
  int wave = threadIdx.x >> 5;
  int n0 = wave * 16;
  bf16x16 bq[4], bk[4], bv[4];
#pragma unroll
  for (int kt = 0; kt < 4; ++kt) {
    const bf16* wrow = Wqkv + (kt * 32) * 384;
    bq[kt] = load_b(wrow + n0,        384, lane);
    bk[kt] = load_b(wrow + 128 + n0,  384, lane);
    bv[kt] = load_b(wrow + 256 + n0,  384, lane);
  }
  int mt0 = blockIdx.x * 8;
  for (int mt = mt0; mt < mt0 + 8; ++mt) {
    long long row0 = (long long)mt * 16;
    __builtin_prefetch(A + (row0 + 16) * DD, 0, 1);   // next A tile -> global_prefetch
    bf16x16 afr[4];
#pragma unroll
    for (int kt = 0; kt < 4; ++kt)
      afr[kt] = load_a(A + row0 * DD + kt * 32, DD, lane);
    floatx8 aq = {}, ak = {}, av = {};
#pragma unroll
    for (int kt = 0; kt < 4; ++kt) {
      aq = wmma_acc(afr[kt], bq[kt], aq);
      ak = wmma_acc(afr[kt], bk[kt], ak);
      av = wmma_acc(afr[kt], bv[kt], av);
    }
    int col = n0 + (lane & 15);
    int rb  = (lane & 16) ? 8 : 0;
#pragma unroll
    for (int i = 0; i < 8; ++i) {
      long long r = (row0 + rb + i) * DD + col;
      q[r]  = (bf16)aq[i];
      kk[r] = (bf16)ak[i];
      v[r]  = (bf16)av[i];
    }
  }
}

// ---------------------------------------------------------------- generic K=128 GEMM
// C(M x 128) = A_bf16(M x 128) @ W_bf16(128 x 128) [+resid] -> outF / outB.
// Compile-time output mode: straight-line epilogue, no per-element branching.

template <bool RESID, bool OUTF, bool OUTB>
__global__ void __launch_bounds__(256, 1)
gemm128_t(const bf16* __restrict__ A, const bf16* __restrict__ W,
          const float* __restrict__ resid, float* __restrict__ outF,
          bf16* __restrict__ outB) {
  int lane = threadIdx.x & 31;
  int wave = threadIdx.x >> 5;
  int n0 = wave * 16;
  bf16x16 bfr[4];
#pragma unroll
  for (int kt = 0; kt < 4; ++kt)
    bfr[kt] = load_b(W + (kt * 32) * DD + n0, DD, lane);
  int mt0 = blockIdx.x * 8;
  for (int mt = mt0; mt < mt0 + 8; ++mt) {
    long long row0 = (long long)mt * 16;
    __builtin_prefetch(A + (row0 + 16) * DD, 0, 1);
    floatx8 acc = {};
#pragma unroll
    for (int kt = 0; kt < 4; ++kt) {
      bf16x16 afr = load_a(A + row0 * DD + kt * 32, DD, lane);
      acc = wmma_acc(afr, bfr[kt], acc);
    }
    int col = n0 + (lane & 15);
    int rb  = (lane & 16) ? 8 : 0;
#pragma unroll
    for (int i = 0; i < 8; ++i) {
      long long r = (row0 + rb + i) * DD + col;
      float vv = acc[i];
      if constexpr (RESID) vv += resid[r];
      if constexpr (OUTF)  outF[r] = vv;
      if constexpr (OUTB)  outB[r] = (bf16)vv;
    }
  }
}

// ---------------------------------------------------------------- fused attention
// One block per batch element. Scores (144x160 f32) live in dynamic LDS.

__global__ void __launch_bounds__(256, 1)
attention_kernel(const bf16* __restrict__ q, const bf16* __restrict__ k,
                 const bf16* __restrict__ v, const float* __restrict__ inp,
                 bf16* __restrict__ mh) {
  extern __shared__ float sc[];            // [144][160]
  const int LDSC = 160;
  int b = blockIdx.x;
  int lane = threadIdx.x & 31;
  int wave = threadIdx.x >> 5;             // 8 waves
  long long base = (long long)b * NTOK;
  const bf16* qb = q + base * DD;
  const bf16* kb = k + base * DD;
  const bf16* vb = v + base * DD;

  // scores = q @ k^T
  for (int mt = wave; mt < 9; mt += 8) {
    for (int nt = 0; nt < 9; ++nt) {
      floatx8 acc = {};
#pragma unroll
      for (int kt = 0; kt < 4; ++kt) {
        bf16x16 afr = load_a (qb + (mt * 16) * DD + kt * 32, DD, lane);
        bf16x16 bfr = load_bt(kb + (nt * 16) * DD + kt * 32, DD, lane);
        acc = wmma_acc(afr, bfr, acc);
      }
      int col = nt * 16 + (lane & 15);
      int rb  = (lane & 16) ? 8 : 0;
#pragma unroll
      for (int i = 0; i < 8; ++i)
        sc[(mt * 16 + rb + i) * LDSC + col] = acc[i];
    }
  }
  __syncthreads();

  // masked softmax, one wave per row (wave32 shuffles)
  for (int r = wave; r < NTOK; r += 8) {
    float mx = -1e30f;
    for (int c = lane; c < NTOK; c += 32) {
      float vd = inp[(base + c) * 9 + 8];
      float x = (vd < 0.5f) ? -1e9f : sc[r * LDSC + c] * INV_SQRT_D;
      sc[r * LDSC + c] = x;
      mx = fmaxf(mx, x);
    }
#pragma unroll
    for (int o = 16; o > 0; o >>= 1) mx = fmaxf(mx, __shfl_xor(mx, o, 32));
    float sum = 0.f;
    for (int c = lane; c < LDSC; c += 32) {
      float e = 0.f;
      if (c < NTOK) e = __expf(sc[r * LDSC + c] - mx);
      sc[r * LDSC + c] = e;                // padding cols become exactly 0
      sum += e;
    }
#pragma unroll
    for (int o = 16; o > 0; o >>= 1) sum += __shfl_xor(sum, o, 32);
    float inv = 1.f / sum;
    for (int c = lane; c < LDSC; c += 32) sc[r * LDSC + c] *= inv;
  }
  __syncthreads();

  // mh = attn @ v (attn read from LDS f32 -> bf16 fragments; K padded to 160)
  for (int mt = wave; mt < 9; mt += 8) {
    for (int nt = 0; nt < 8; ++nt) {
      floatx8 acc = {};
#pragma unroll
      for (int kt = 0; kt < 5; ++kt) {
        bf16x16 afr = load_a_f32(sc + (mt * 16) * LDSC + kt * 32, LDSC, lane);
        bf16x16 bfr = load_b(vb + (kt * 32) * DD + nt * 16, DD, lane);
        acc = wmma_acc(afr, bfr, acc);
      }
      int col = nt * 16 + (lane & 15);
      int rb  = (lane & 16) ? 8 : 0;
#pragma unroll
      for (int i = 0; i < 8; ++i) {
        int r = mt * 16 + rb + i;
        if (r < NTOK) mh[(base + r) * DD + col] = (bf16)acc[i];
      }
    }
  }
}

// ---------------------------------------------------------------- fused FF block
// One block per 16-row tile: h += relu(X@W1 + b1)@W2 + b2. X and hidden in LDS.

__global__ void __launch_bounds__(256, 1)
ffn_kernel(const bf16* __restrict__ X, const bf16* __restrict__ W1,
           const float* __restrict__ b1, const bf16* __restrict__ W2,
           const float* __restrict__ b2, float* __restrict__ h) {
  __shared__ __align__(16) bf16 xt[16 * DD];
  __shared__ __align__(16) bf16 hid[16 * FFH];
  long long row0 = (long long)blockIdx.x * 16;
  int lane = threadIdx.x & 31;
  int wave = threadIdx.x >> 5;
  for (int i = threadIdx.x; i < 16 * DD; i += 256) xt[i] = X[row0 * DD + i];
  __syncthreads();
  // hidden = relu(X @ W1 + b1)
  for (int nt = wave; nt < FFH / 16; nt += 8) {
    floatx8 acc = {};
#pragma unroll
    for (int kt = 0; kt < 4; ++kt) {
      bf16x16 afr = load_a(xt + kt * 32, DD, lane);
      bf16x16 bfr = load_b(W1 + (kt * 32) * FFH + nt * 16, FFH, lane);
      acc = wmma_acc(afr, bfr, acc);
    }
    int col = nt * 16 + (lane & 15);
    int rb  = (lane & 16) ? 8 : 0;
#pragma unroll
    for (int i = 0; i < 8; ++i) {
      float vv = fmaxf(acc[i] + b1[col], 0.f);
      hid[(rb + i) * FFH + col] = (bf16)vv;
    }
  }
  __syncthreads();
  // h += hidden @ W2 + b2
  {
    int nt = wave;                         // 8 waves, 8 N-tiles
    floatx8 acc = {};
    for (int kt = 0; kt < FFH / 32; ++kt) {
      bf16x16 afr = load_a(hid + kt * 32, FFH, lane);
      bf16x16 bfr = load_b(W2 + (kt * 32) * DD + nt * 16, DD, lane);
      acc = wmma_acc(afr, bfr, acc);
    }
    int col = nt * 16 + (lane & 15);
    int rb  = (lane & 16) ? 8 : 0;
#pragma unroll
    for (int i = 0; i < 8; ++i) {
      long long r = row0 + rb + i;
      h[r * DD + col] += acc[i] + b2[col];
    }
  }
}

// ---------------------------------------------------------------- batch-norm

__global__ void zero_stats_kernel(float* s) { s[threadIdx.x] = 0.f; }   // <<<1,256>>>

__global__ void zero_pads_kernel(bf16* qp, bf16* kp, bf16* vp) {
  int i = blockIdx.x * blockDim.x + threadIdx.x;
  if (i < 32 * DD) { qp[i] = (bf16)0.f; kp[i] = (bf16)0.f; vp[i] = (bf16)0.f; }
}

__global__ void bn_stats_kernel(const float* __restrict__ x, float* __restrict__ stats) {
  int c = threadIdx.x;                      // blockDim = 128 channels
  float s = 0.f, s2 = 0.f;
  for (long long r = blockIdx.x; r < MR; r += gridDim.x) {
    float vv = x[r * DD + c];
    s += vv; s2 += vv * vv;
  }
  atomicAdd(&stats[c], s);
  atomicAdd(&stats[DD + c], s2);
}

__global__ void bn_apply_kernel(float* __restrict__ h, bf16* __restrict__ hb,
                                const float* __restrict__ stats,
                                const float* __restrict__ g, const float* __restrict__ bb) {
  const float invM = 1.f / (float)MR;
  long long total = (long long)MR * DD;
  for (long long idx = (long long)blockIdx.x * blockDim.x + threadIdx.x; idx < total;
       idx += (long long)gridDim.x * blockDim.x) {
    int c = (int)(idx & (DD - 1));
    float mu  = stats[c] * invM;
    float var = stats[DD + c] * invM - mu * mu;
    float y = (h[idx] - mu) * rsqrtf(var + 1e-5f) * g[c] + bb[c];
    h[idx]  = y;
    hb[idx] = (bf16)y;
  }
}

// ---------------------------------------------------------------- graph embed + head

__global__ void graph_embed_kernel(const float* __restrict__ h, const float* __restrict__ inp,
                                   bf16* __restrict__ geb) {
  int b = blockIdx.x, c = threadIdx.x;      // blockDim = 128
  long long base = (long long)b * NTOK;
  float s = 0.f, vl = 0.f;
  for (int n = 0; n < NTOK; ++n) {
    float vd = inp[(base + n) * 9 + 8];
    vl += vd;
    if (vd >= 0.5f) s += h[(base + n) * DD + c];
  }
  geb[b * DD + c] = (bf16)(s / vl);
}

__global__ void head_kernel(const float* __restrict__ h, const float* __restrict__ inp,
                            const float* __restrict__ wnode, const float* __restrict__ ctx,
                            float* __restrict__ alp, float* __restrict__ sel,
                            float* __restrict__ ent, float* __restrict__ probs) {
  __shared__ float cx[DD];
  __shared__ float wv[DD];
  __shared__ float lg[50];
  int b = blockIdx.x, t = threadIdx.x;      // blockDim = 128
  long long base = (long long)b * NTOK;
  cx[t] = ctx[b * DD + t];
  __syncthreads();
  {
    // wv = W_node[:, :D] @ ctx  (so comp_n = emb_n . wv)
    const float* wr = wnode + t * 384;
    float s = 0.f;
#pragma unroll 8
    for (int d = 0; d < DD; ++d) s += wr[d] * cx[d];
    wv[t] = s;
  }
  __syncthreads();
  if (t < 50) {
    int tok = 80 + t;
    float vd = inp[(base + tok) * 9 + 8];
    const float* hr = h + (base + tok) * DD;
    float s = 0.f;
#pragma unroll 8
    for (int d = 0; d < DD; ++d) s += hr[d] * wv[d];
    if (vd < 0.5f) s = 0.f;                 // emb masked to zero for invalid tokens
    lg[t] = tanhf(s * INV_SQRT_D) * 10.f;
  }
  __syncthreads();
  if (t == 0) {
    float mx = -1e30f;
    for (int i = 0; i < 50; ++i) mx = fmaxf(mx, lg[i]);
    float p[50]; float Z = 0.f;
    for (int i = 0; i < 50; ++i) { p[i] = __expf(lg[i] - mx); Z += p[i]; }
    float S2 = 0.f;
    for (int i = 0; i < 50; ++i) {
      float vd = inp[(base + 80 + i) * 9 + 8];
      p[i] = (vd < 0.5f) ? 0.f : p[i] / Z;  // mask by leaf validity
      S2 += p[i];
    }
    float inv = 1.f / S2;
    float e = 0.f; int am = 0; float bm = -1.f;
    for (int i = 0; i < 50; ++i) {
      float qv = p[i] * inv;
      probs[b * 50 + i] = qv;
      if (qv > 0.f) e -= qv * __logf(fmaxf(qv, 1e-38f));
      if (qv > bm) { bm = qv; am = i; }     // strict > keeps first max (jnp.argmax)
    }
    ent[b] = e;
    sel[b] = (float)am;
    alp[b] = __logf(fmaxf(bm, 1e-38f));
  }
}

// ---------------------------------------------------------------- launch

extern "C" void kernel_launch(void* const* d_in, const int* in_sizes, int n_in,
                              void* d_out, int out_size, void* d_ws, size_t ws_size,
                              hipStream_t stream) {
  (void)in_sizes; (void)n_in; (void)out_size; (void)ws_size;
  const float* inp   = (const float*)d_in[0];
  const float* wnode = (const float*)d_in[13];
  const float* wctx  = (const float*)d_in[14];

  char* ws = (char*)d_ws;
  size_t off = 0;
  auto carve = [&](size_t bytes) -> void* {
    void* p = ws + off;
    off = (off + bytes + 255) & ~(size_t)255;
    return p;
  };
  float* h    = (float*)carve((size_t)MR * DD * 4);
  bf16*  hb   = (bf16*) carve((size_t)MR * DD * 2);
  bf16*  q    = (bf16*) carve((size_t)MPAD * DD * 2);
  bf16*  kk   = (bf16*) carve((size_t)MPAD * DD * 2);
  bf16*  v    = (bf16*) carve((size_t)MPAD * DD * 2);
  bf16*  mh   = (bf16*) carve((size_t)MR * DD * 2);
  bf16*  ge   = (bf16*) carve((size_t)NB * DD * 2);
  float* stats = (float*)carve(256 * 4);
  bf16*  wbf  = (bf16*) carve((size_t)409600 * 2);

  // per-layer bf16 weight layout: [wqkv 128x384][wo 128x128][w1 128x512][w2 512x128]
  bf16 *wqkv_b[2], *wo_b[2], *w1_b[2], *w2_b[2];
  for (int l = 0; l < 2; ++l) {
    bf16* base = wbf + (size_t)l * 196608;
    wqkv_b[l] = base;
    wo_b[l]   = base + 49152;
    w1_b[l]   = base + 65536;
    w2_b[l]   = base + 131072;
  }
  bf16* wctx_b = wbf + 393216;

  CvtJobs jobs;
  int ji = 0;
  for (int l = 0; l < 2; ++l) {
    const int p0 = 15 + 12 * l;
    jobs.j[ji++] = { (const float*)d_in[p0 + 0], wqkv_b[l], 128, 128, 384,   0 };
    jobs.j[ji++] = { (const float*)d_in[p0 + 1], wqkv_b[l], 128, 128, 384, 128 };
    jobs.j[ji++] = { (const float*)d_in[p0 + 2], wqkv_b[l], 128, 128, 384, 256 };
    jobs.j[ji++] = { (const float*)d_in[p0 + 3], wo_b[l],   128, 128, 128,   0 };
    jobs.j[ji++] = { (const float*)d_in[p0 + 4], w1_b[l],   128, 512, 512,   0 };
    jobs.j[ji++] = { (const float*)d_in[p0 + 6], w2_b[l],   512, 128, 128,   0 };
  }
  jobs.j[ji++] = { wctx, wctx_b, 128, 128, 128, 0 };
  cvt_kernel<<<256, 256, 0, stream>>>(jobs);

  EmbedW ew = { (const float*)d_in[1],  (const float*)d_in[2],  (const float*)d_in[3],  (const float*)d_in[4],
                (const float*)d_in[5],  (const float*)d_in[6],  (const float*)d_in[7],  (const float*)d_in[8],
                (const float*)d_in[9],  (const float*)d_in[10], (const float*)d_in[11], (const float*)d_in[12] };
  embed_kernel<<<MR, 128, 0, stream>>>(inp, ew, h, hb);
  zero_pads_kernel<<<16, 256, 0, stream>>>(q + (size_t)MR * DD, kk + (size_t)MR * DD,
                                           v + (size_t)MR * DD);

  for (int l = 0; l < 2; ++l) {
    const int p0 = 15 + 12 * l;
    const float* ffb1 = (const float*)d_in[p0 + 5];
    const float* ffb2 = (const float*)d_in[p0 + 7];
    const float* n1g  = (const float*)d_in[p0 + 8];
    const float* n1b  = (const float*)d_in[p0 + 9];
    const float* n2g  = (const float*)d_in[p0 + 10];
    const float* n2b  = (const float*)d_in[p0 + 11];

    gemm_qkv<<<MR / 128, 256, 0, stream>>>(hb, wqkv_b[l], q, kk, v);
    attention_kernel<<<NB, 256, 144 * 160 * 4, stream>>>(q, kk, v, inp, mh);
    gemm128_t<true, true, false><<<MR / 128, 256, 0, stream>>>(mh, wo_b[l], h, h,
                                                               nullptr);  // +residual
    zero_stats_kernel<<<1, 256, 0, stream>>>(stats);
    bn_stats_kernel<<<256, 128, 0, stream>>>(h, stats);
    bn_apply_kernel<<<4096, 256, 0, stream>>>(h, hb, stats, n1g, n1b);
    ffn_kernel<<<MR / 16, 256, 0, stream>>>(hb, w1_b[l], ffb1, w2_b[l], ffb2, h);
    zero_stats_kernel<<<1, 256, 0, stream>>>(stats);
    bn_stats_kernel<<<256, 128, 0, stream>>>(h, stats);
    bn_apply_kernel<<<4096, 256, 0, stream>>>(h, hb, stats, n2g, n2b);
  }

  float* out   = (float*)d_out;
  float* alp   = out;
  float* sel   = out + NB;
  float* ent   = out + 2 * NB;
  float* ctx   = out + 3 * NB;              // 1024 x 128
  float* probs = out + 3 * NB + NB * DD;    // 1024 x 50

  graph_embed_kernel<<<NB, 128, 0, stream>>>(h, inp, ge);
  gemm128_t<false, true, false><<<NB / 128, 256, 0, stream>>>(ge, wctx_b, nullptr,
                                                              ctx, nullptr);
  head_kernel<<<NB, 128, 0, stream>>>(h, inp, wnode, ctx, alp, sel, ent, probs);
}
